// Encoder_40132174414075
// MI455X (gfx1250) — matrix operand order repaired and checked by reading, not compiled, not run on hardware
//
#include <hip/hip_runtime.h>
#include <hip/hip_bf16.h>

typedef __attribute__((ext_vector_type(16))) _Float16 v16h;
typedef __attribute__((ext_vector_type(8)))  _Float16 v8h;
typedef __attribute__((ext_vector_type(8)))  float    v8f;

#define N_ROWS 16384      // B*S
#define D_K    1280       // D_IN == TOKEN_DIM
#define K_EMB  8192       // TOKEN_SIZE

union V16 { v16h v; struct { v8h lo, hi; } p; };

// A fragment: 16x32 f16, row-major source with leading dim ld.
// lane&15 = M row, lane>>4 selects K sub-chunks {0..7,16..23} vs {8..15,24..31}
__device__ inline v16h load_frag_a(const _Float16* src, int ld, int m, int hi, int kBase) {
    V16 u;
    const _Float16* p = src + (size_t)m * ld + kBase + hi * 8;
    u.p.lo = *(const v8h*)(p);
    u.p.hi = *(const v8h*)(p + 16);
    return u.v;
}

// B fragment: 32x16 f16. Source row-major [col, K]. lane&15 = N col,
// lane>>4 selects K sub-chunks {0..7,16..23} vs {8..15,24..31}
__device__ inline v16h load_frag_b(const _Float16* src, int ld, int n, int hi, int kBase) {
    V16 u;
    const _Float16* p = src + (size_t)n * ld + kBase + hi * 8;
    u.p.lo = *(const v8h*)(p);
    u.p.hi = *(const v8h*)(p + 16);
    return u.v;
}

__global__ void cvt_f16(const float* __restrict__ in, _Float16* __restrict__ out, int n) {
    int i = blockIdx.x * 256 + threadIdx.x;
    if (i < n) out[i] = (_Float16)in[i];
}

// Convert embedding row to f16 and compute ||e||^2 (f32) in one pass.
__global__ void embed_prep(const float* __restrict__ emb, _Float16* __restrict__ embH,
                           float* __restrict__ e2) {
    const int k = blockIdx.x;
    const float* row = emb + (size_t)k * D_K;
    _Float16* rowH = embH + (size_t)k * D_K;
    float s = 0.f;
    for (int d = threadIdx.x; d < D_K; d += 128) {
        float v = row[d];
        rowH[d] = (_Float16)v;
        s += v * v;
    }
    for (int mask = 16; mask >= 1; mask >>= 1) s += __shfl_xor(s, mask, 32);
    __shared__ float wsum[4];
    if ((threadIdx.x & 31) == 0) wsum[threadIdx.x >> 5] = s;
    __syncthreads();
    if (threadIdx.x == 0) e2[k] = wsum[0] + wsum[1] + wsum[2] + wsum[3];
}

// GEMM1: imgF[n,t] = feat[n,:] . li_w[t,:] + li_b[t]   (16384 x 1280, K=1280)
// One wave per 16x64 tile. Fully unrolled K, grouped 2 k-steps per group:
// all 20 b128 loads of a group are issued before its 8 WMMAs so several
// loads stay in flight. launch_bounds(,2) relaxes the VGPR budget so the
// scheduler doesn't recycle fragment registers immediately.
__global__ void __launch_bounds__(256, 2) gemm1_wmma(
        const _Float16* __restrict__ featH, const _Float16* __restrict__ liwH,
        const float* __restrict__ lib, float* __restrict__ imgF,
        _Float16* __restrict__ imgH) {
    const int lane = threadIdx.x & 31;
    const int wave = threadIdx.x >> 5;
    const int gw = blockIdx.x * 8 + wave;          // 20480 wave tiles
    const int rowBase = (gw / 20) * 16;
    const int colBase = (gw % 20) * 64;
    const int m = lane & 15;
    const int hi = lane >> 4;

    const _Float16* aSrc = featH + (size_t)(rowBase + m) * D_K;
    const _Float16* bSrc = liwH + (size_t)(colBase + m) * D_K;

    v8f acc[4] = {};
#pragma unroll
    for (int kBase = 0; kBase < D_K; kBase += 64) {
        // issue all loads of this 64-wide K group first
        v16h a0 = load_frag_a(aSrc, D_K, 0, hi, kBase);
        v16h a1 = load_frag_a(aSrc, D_K, 0, hi, kBase + 32);
        v16h b0[4], b1[4];
#pragma unroll
        for (int j = 0; j < 4; ++j)
            b0[j] = load_frag_b(bSrc + (size_t)j * 16 * D_K, D_K, 0, hi, kBase);
#pragma unroll
        for (int j = 0; j < 4; ++j)
            b1[j] = load_frag_b(bSrc + (size_t)j * 16 * D_K, D_K, 0, hi, kBase + 32);
        // then the 8 WMMAs
#pragma unroll
        for (int j = 0; j < 4; ++j)
            acc[j] = __builtin_amdgcn_wmma_f32_16x16x32_f16(
                false, a0, false, b0[j], (short)0, acc[j], false, false);
#pragma unroll
        for (int j = 0; j < 4; ++j)
            acc[j] = __builtin_amdgcn_wmma_f32_16x16x32_f16(
                false, a1, false, b1[j], (short)0, acc[j], false, false);
    }

#pragma unroll
    for (int j = 0; j < 4; ++j) {
        const int col = colBase + j * 16 + m;
        const float bias = lib[col];
#pragma unroll
        for (int r = 0; r < 8; ++r) {
            const int row = rowBase + r + 8 * hi;
            float v = acc[j][r] + bias;
            imgF[(size_t)row * D_K + col] = v;
            imgH[(size_t)row * D_K + col] = (_Float16)v;
        }
    }
}

// GEMM2 + fused argmin: score[n,k] = ||e_k||^2 - 2 * x_n . e_k
// Block = 32 rows: two 16-row A strips staged in LDS (80KB) via async-to-LDS,
// 8 waves x 1024 columns. Each B fragment feeds 2 WMMAs. K loop in 64-wide
// groups (loads first, then 4 WMMAs), unrolled x4.
__global__ void __launch_bounds__(256, 2) gemm2_argmin(
        const _Float16* __restrict__ imgH, const _Float16* __restrict__ embH,
        const float* __restrict__ e2, int* __restrict__ best) {
    __shared__ __align__(32) _Float16 As[32 * D_K];   // 80 KB
    __shared__ float redV[8][32];
    __shared__ int   redI[8][32];

    const int rowBase = blockIdx.x * 32;
    {   // async 80KB contiguous fill: 32 consecutive rows of imgH -> LDS
        const _Float16* gsrc = imgH + (size_t)rowBase * D_K;
        for (int i = threadIdx.x; i < (32 * D_K) / 8; i += 256) {      // b128 chunks
            unsigned ldsOff = (unsigned)(uintptr_t)(&As[i * 8]);
            unsigned long long ga = (unsigned long long)(uintptr_t)(gsrc + (size_t)i * 8);
            asm volatile("global_load_async_to_lds_b128 %0, %1, off"
                         :: "v"(ldsOff), "v"(ga) : "memory");
        }
        asm volatile("s_wait_asynccnt 0x0" ::: "memory");
    }
    __syncthreads();

    const int lane = threadIdx.x & 31;
    const int wave = threadIdx.x >> 5;
    const int m = lane & 15;
    const int hi = lane >> 4;
    const int colStart = wave * (K_EMB / 8);

    const _Float16* a0Src = As + (size_t)m * D_K;              // strip 0 row
    const _Float16* a1Src = As + (size_t)(16 + m) * D_K;       // strip 1 row

    float bestVal[2][8];
    int bestIdx[2][8];
#pragma unroll
    for (int s = 0; s < 2; ++s)
#pragma unroll
        for (int r = 0; r < 8; ++r) { bestVal[s][r] = 3.4e38f; bestIdx[s][r] = 0; }

    for (int ct = 0; ct < (K_EMB / 8) / 16; ++ct) {
        const int colBase = colStart + ct * 16;
        if (ct + 1 < (K_EMB / 8) / 16)   // prefetch next tile's embedding row
            __builtin_prefetch(embH + (size_t)(colBase + 16 + m) * D_K, 0, 1);

        const _Float16* bSrc = embH + (size_t)(colBase + m) * D_K;

        v8f acc0 = {}, acc1 = {};
#pragma unroll 4
        for (int kBase = 0; kBase < D_K; kBase += 64) {
            // loads of the 64-wide K group first (global B + LDS A)
            v16h bA = load_frag_b(bSrc, D_K, 0, hi, kBase);
            v16h bB = load_frag_b(bSrc, D_K, 0, hi, kBase + 32);
            v16h x0A = load_frag_a(a0Src, D_K, 0, hi, kBase);        // ds_load
            v16h x1A = load_frag_a(a1Src, D_K, 0, hi, kBase);        // ds_load
            v16h x0B = load_frag_a(a0Src, D_K, 0, hi, kBase + 32);
            v16h x1B = load_frag_a(a1Src, D_K, 0, hi, kBase + 32);
            // then 4 WMMAs
            acc0 = __builtin_amdgcn_wmma_f32_16x16x32_f16(
                false, x0A, false, bA, (short)0, acc0, false, false);
            acc1 = __builtin_amdgcn_wmma_f32_16x16x32_f16(
                false, x1A, false, bA, (short)0, acc1, false, false);
            acc0 = __builtin_amdgcn_wmma_f32_16x16x32_f16(
                false, x0B, false, bB, (short)0, acc0, false, false);
            acc1 = __builtin_amdgcn_wmma_f32_16x16x32_f16(
                false, x1B, false, bB, (short)0, acc1, false, false);
        }

        const float e2c = e2[colBase + m];
        v8f accs[2];
        accs[0] = acc0; accs[1] = acc1;
#pragma unroll
        for (int s = 0; s < 2; ++s) {
#pragma unroll
            for (int r = 0; r < 8; ++r) {
                float v = e2c - 2.0f * accs[s][r];  // row = s*16 + r + 8*hi
                int idx = colBase + m;
#pragma unroll
                for (int mask = 1; mask < 16; mask <<= 1) {  // 16-lane butterfly min
                    float ov = __shfl_xor(v, mask, 32);
                    int oi = __shfl_xor(idx, mask, 32);
                    if (ov < v || (ov == v && oi < idx)) { v = ov; idx = oi; }
                }
                if (v < bestVal[s][r] || (v == bestVal[s][r] && idx < bestIdx[s][r])) {
                    bestVal[s][r] = v; bestIdx[s][r] = idx;
                }
            }
        }
    }

    if (m == 0) {
#pragma unroll
        for (int s = 0; s < 2; ++s)
#pragma unroll
            for (int r = 0; r < 8; ++r) {
                redV[wave][s * 16 + hi * 8 + r] = bestVal[s][r];
                redI[wave][s * 16 + hi * 8 + r] = bestIdx[s][r];
            }
    }
    __syncthreads();
    if (wave == 0) {   // 32 lanes reduce 32 rows across 8 waves
        float v = redV[0][lane]; int idx = redI[0][lane];
        for (int w = 1; w < 8; ++w) {
            float ov = redV[w][lane]; int oi = redI[w][lane];
            if (ov < v || (ov == v && oi < idx)) { v = ov; idx = oi; }
        }
        best[rowBase + lane] = idx;
    }
}

__global__ void zero_loss(float* p) { if (threadIdx.x == 0 && blockIdx.x == 0) *p = 0.f; }

// out[n,:] = embedding[best[n],:] (straight-through value); loss += 1.25*mean(diff^2)
__global__ void __launch_bounds__(256) gather_loss(
        const int* __restrict__ best, const float* __restrict__ emb,
        const float* __restrict__ imgF, float* __restrict__ out,
        float* __restrict__ lossAcc) {
    const int n = blockIdx.x;
    const int idx = best[n];
    const float* e = emb + (size_t)idx * D_K;
    const float* x = imgF + (size_t)n * D_K;
    float* o = out + (size_t)n * D_K;
    float s = 0.f;
    for (int d = threadIdx.x; d < D_K; d += 256) {
        float ev = e[d];
        o[d] = ev;
        float diff = x[d] - ev;
        s += diff * diff;
    }
    for (int mask = 16; mask >= 1; mask >>= 1) s += __shfl_xor(s, mask, 32);
    __shared__ float wsum[8];
    if ((threadIdx.x & 31) == 0) wsum[threadIdx.x >> 5] = s;
    __syncthreads();
    if (threadIdx.x == 0) {
        float t = 0.f;
        for (int w = 0; w < 8; ++w) t += wsum[w];
        atomicAdd(lossAcc, t * (1.25f / ((float)N_ROWS * (float)D_K)));
    }
}

extern "C" void kernel_launch(void* const* d_in, const int* in_sizes, int n_in,
                              void* d_out, int out_size, void* d_ws, size_t ws_size,
                              hipStream_t stream) {
    const float* feat = (const float*)d_in[0];   // [64,256,1280]
    const float* liw  = (const float*)d_in[1];   // [1280,1280]
    const float* lib  = (const float*)d_in[2];   // [1280]
    const float* emb  = (const float*)d_in[3];   // [8192,1280]
    float* out = (float*)d_out;                  // 16384*1280 feature + 1 loss

    char* ws = (char*)d_ws;
    size_t off = 0;
    auto alloc = [&](size_t bytes) -> void* {
        void* p = ws + off;
        off = (off + bytes + 255) & ~(size_t)255;
        return p;
    };
    _Float16* featH = (_Float16*)alloc((size_t)N_ROWS * D_K * 2);
    _Float16* liwH  = (_Float16*)alloc((size_t)D_K * D_K * 2);
    _Float16* embH  = (_Float16*)alloc((size_t)K_EMB * D_K * 2);
    float*    e2    = (float*)   alloc((size_t)K_EMB * 4);
    float*    imgF  = (float*)   alloc((size_t)N_ROWS * D_K * 4);
    _Float16* imgH  = (_Float16*)alloc((size_t)N_ROWS * D_K * 2);
    int*      best  = (int*)     alloc((size_t)N_ROWS * 4);

    float* lossPtr = out + (size_t)N_ROWS * D_K;

    cvt_f16<<<(N_ROWS * D_K + 255) / 256, 256, 0, stream>>>(feat, featH, N_ROWS * D_K);
    cvt_f16<<<(D_K * D_K + 255) / 256, 256, 0, stream>>>(liw, liwH, D_K * D_K);
    embed_prep<<<K_EMB, 128, 0, stream>>>(emb, embH, e2);
    gemm1_wmma<<<(N_ROWS / 16) * (D_K / 64) / 8, 256, 0, stream>>>(featH, liwH, lib, imgF, imgH);
    gemm2_argmin<<<N_ROWS / 32, 256, 0, stream>>>(imgH, embH, e2, best);
    zero_loss<<<1, 32, 0, stream>>>(lossPtr);
    gather_loss<<<N_ROWS, 256, 0, stream>>>(best, emb, imgF, out, lossPtr);
}
